// MambaIRv2_52080773431766
// MI455X (gfx1250) — compile-verified, matrix operand order checked by reading, not run on hardware
//
#include <hip/hip_runtime.h>
#include <hip/hip_bf16.h>
#include <math.h>

// ---------------- problem constants ----------------
#define BB      4
#define LL      4096
#define DMODEL  384
#define DSTATE  16
#define DTRANK  24
#define CTOT    (DTRANK + 2*DSTATE)   // 56
#define MTOT    (BB*LL)               // 16384 token rows
#define NCH     32                    // scan chunks per sequence
#define CHL     (LL/NCH)              // 128 steps per chunk

typedef float     v2f  __attribute__((ext_vector_type(2)));
typedef float     v8f  __attribute__((ext_vector_type(8)));
typedef _Float16  v16h __attribute__((ext_vector_type(16)));

#if defined(__gfx1250__) && __has_builtin(__builtin_amdgcn_wmma_f32_16x16x4_f32)
#define HAVE_WMMA_F32 1
#else
#define HAVE_WMMA_F32 0
#endif

// =====================================================================
// Kernel 1: projection GEMM  out[m,c] = sum_k X[m,k] * W1[c,k]
//   M = 16384, N = 56 (padded to 64), K = 384.
//   grid.x = M/128 (8 waves * 16 rows), grid.y = 4 (N tile, c0 = 16*y).
//   W1 N-slice staged in LDS with stride 388 (388%64==4 -> conflict-free
//   across the 16-lane half-wave). Epilogue routes c<24 -> dts_r,
//   24..39 -> Bs, 40..55 -> Cs (+prompt fused).
// =====================================================================
__global__ __launch_bounds__(256)
void proj_gemm_kernel(const float* __restrict__ X,
                      const float* __restrict__ W1,
                      const float* __restrict__ prompt,
                      float* __restrict__ dtsr,
                      float* __restrict__ Bsc,
                      float* __restrict__ Csc)
{
    __shared__ float Wlds[16 * 388];

    const int c0 = blockIdx.y * 16;
    // stage 16 rows of W1 (zero-padded past c=55)
    for (int i = threadIdx.x; i < 16 * DMODEL; i += 256) {
        int n = i / DMODEL, k = i - n * DMODEL;
        float w = (c0 + n < CTOT) ? W1[(c0 + n) * DMODEL + k] : 0.0f;
        Wlds[n * 388 + k] = w;
    }
    __syncthreads();

    const int wave = threadIdx.x >> 5;
    const int lane = threadIdx.x & 31;
    const int ln   = lane & 15;        // A: row M ; B: col N ; C/D: col N
    const int hi   = lane >> 4;        // K-half select for A/B frags
    const int m0   = blockIdx.x * 128 + wave * 16;

    v8f acc = {};
    const float* arow = X + (size_t)(m0 + ln) * DMODEL;

#if HAVE_WMMA_F32
    for (int k0 = 0; k0 < DMODEL; k0 += 4) {
        v2f a, b;
        a.x = arow[k0 + 2 * hi];
        a.y = arow[k0 + 2 * hi + 1];
        b.x = Wlds[ln * 388 + k0 + 2 * hi];
        b.y = Wlds[ln * 388 + k0 + 2 * hi + 1];
        acc = __builtin_amdgcn_wmma_f32_16x16x4_f32(
                  false, a, false, b, (short)0, acc, false, false);
    }
#else
    for (int k0 = 0; k0 < DMODEL; k0 += 32) {
        v16h a, b;
#pragma unroll
        for (int v = 0; v < 8; ++v) {
            int kk = (v < 4 ? 2 * v : 16 + 2 * (v - 4)) + 8 * hi; // ISA 16b A layout
            a[2 * v]     = (_Float16)arow[k0 + kk];
            a[2 * v + 1] = (_Float16)arow[k0 + kk + 1];
            b[2 * v]     = (_Float16)Wlds[ln * 388 + k0 + kk];
            b[2 * v + 1] = (_Float16)Wlds[ln * 388 + k0 + kk + 1];
        }
        acc = __builtin_amdgcn_wmma_f32_16x16x32_f16(
                  false, a, false, b, (short)0, acc, false, false);
    }
#endif

    // C/D layout: col N = ln, row M = r + 8*hi
    const int c = c0 + ln;
#pragma unroll
    for (int r = 0; r < 8; ++r) {
        int m = m0 + r + 8 * hi;
        float v = acc[r];
        if (c < DTRANK) {
            dtsr[(size_t)m * DTRANK + c] = v;
        } else if (c < DTRANK + DSTATE) {
            Bsc[(size_t)m * DSTATE + (c - DTRANK)] = v;
        } else if (c < CTOT) {
            int n = c - DTRANK - DSTATE;
            Csc[(size_t)m * DSTATE + n] = v + prompt[(size_t)m * DSTATE + n];
        }
    }
}

// =====================================================================
// Kernel 2: dt GEMM + bias + softplus
//   delta[m,d] = softplus( sum_r dtsr[m,r]*W2[d,r] + bias[d] )
//   M = 16384, N = 384 (24 tiles), K = 24 (6 f32 WMMA k-steps).
//   A fragments for the whole K live in registers (12 floats/lane).
// =====================================================================
__global__ __launch_bounds__(256)
void dt_gemm_kernel(const float* __restrict__ dtsr,
                    const float* __restrict__ W2,
                    const float* __restrict__ bias,
                    float* __restrict__ delta)
{
    __shared__ float Wlds[DMODEL * 28];   // 43 KB, stride 28 -> conflict-free

    for (int i = threadIdx.x; i < DMODEL * DTRANK; i += 256) {
        int d = i / DTRANK, r = i - d * DTRANK;
        Wlds[d * 28 + r] = W2[i];
    }
    __syncthreads();

    const int wave = threadIdx.x >> 5;
    const int lane = threadIdx.x & 31;
    const int ln   = lane & 15;
    const int hi   = lane >> 4;
    const int m0   = blockIdx.x * 128 + wave * 16;

    const float* arow = dtsr + (size_t)(m0 + ln) * DTRANK;

#if HAVE_WMMA_F32
    v2f afr[6];
#pragma unroll
    for (int ks = 0; ks < 6; ++ks) {
        afr[ks].x = arow[4 * ks + 2 * hi];
        afr[ks].y = arow[4 * ks + 2 * hi + 1];
    }
#else
    v16h afr;
#pragma unroll
    for (int v = 0; v < 8; ++v) {
        int kk = (v < 4 ? 2 * v : 16 + 2 * (v - 4)) + 8 * hi;
        afr[2 * v]     = (kk     < DTRANK) ? (_Float16)arow[kk]     : (_Float16)0.f;
        afr[2 * v + 1] = (kk + 1 < DTRANK) ? (_Float16)arow[kk + 1] : (_Float16)0.f;
    }
#endif

    for (int d0 = 0; d0 < DMODEL; d0 += 16) {
        v8f acc = {};
#if HAVE_WMMA_F32
#pragma unroll
        for (int ks = 0; ks < 6; ++ks) {
            v2f b;
            b.x = Wlds[(d0 + ln) * 28 + 4 * ks + 2 * hi];
            b.y = Wlds[(d0 + ln) * 28 + 4 * ks + 2 * hi + 1];
            acc = __builtin_amdgcn_wmma_f32_16x16x4_f32(
                      false, afr[ks], false, b, (short)0, acc, false, false);
        }
#else
        v16h b;
#pragma unroll
        for (int v = 0; v < 8; ++v) {
            int kk = (v < 4 ? 2 * v : 16 + 2 * (v - 4)) + 8 * hi;
            b[2 * v]     = (kk     < DTRANK) ? (_Float16)Wlds[(d0 + ln) * 28 + kk]     : (_Float16)0.f;
            b[2 * v + 1] = (kk + 1 < DTRANK) ? (_Float16)Wlds[(d0 + ln) * 28 + kk + 1] : (_Float16)0.f;
        }
        acc = __builtin_amdgcn_wmma_f32_16x16x32_f16(
                  false, afr, false, b, (short)0, acc, false, false);
#endif
        const int   d  = d0 + ln;
        const float bi = bias[d];
#pragma unroll
        for (int r = 0; r < 8; ++r) {
            int   m = m0 + r + 8 * hi;
            float t = acc[r] + bi;
            float sp = (t > 20.0f) ? t : log1pf(expf(t));   // softplus
            delta[(size_t)m * DMODEL + d] = sp;
        }
    }
}

// =====================================================================
// Scan pass A: per (chunk, b, d) compute local state h_end (from h=0)
// and cumulative per-state decay product P = prod exp(dt*A[n]).
// One thread per unit; lanes index consecutive d -> coalesced delta/x.
// =====================================================================
__global__ __launch_bounds__(256)
void scan_passA_kernel(const float* __restrict__ delta,
                       const float* __restrict__ X,
                       const float* __restrict__ Bsc,
                       const float* __restrict__ Alog,
                       float* __restrict__ Pout,
                       float* __restrict__ Hend)
{
    const int idx = blockIdx.x * 256 + threadIdx.x;     // [0, NCH*BB*DMODEL)
    const int d   = idx % DMODEL;
    const int t2  = idx / DMODEL;
    const int b   = t2 % BB;
    const int ch  = t2 / BB;

    float A[DSTATE], h[DSTATE], P[DSTATE];
#pragma unroll
    for (int n = 0; n < DSTATE; ++n) {
        A[n] = -expf(Alog[d * DSTATE + n]);
        h[n] = 0.0f;
        P[n] = 1.0f;
    }

    const int l0 = ch * CHL;
    for (int t = 0; t < CHL; ++t) {
        const int bl = b * LL + l0 + t;
        if (t + 8 < CHL) {  // pull next lines into WGP$ (global_prefetch_b8)
            __builtin_prefetch(&delta[(size_t)(bl + 8) * DMODEL + d], 0, 1);
            __builtin_prefetch(&X[(size_t)(bl + 8) * DMODEL + d], 0, 1);
        }
        const float dt = delta[(size_t)bl * DMODEL + d];
        const float u  = X[(size_t)bl * DMODEL + d];
        const float du = dt * u;
        const float4* Bv4 = reinterpret_cast<const float4*>(Bsc + (size_t)bl * DSTATE);
        float Bl[DSTATE];
#pragma unroll
        for (int q = 0; q < 4; ++q) {
            float4 v = Bv4[q];
            Bl[4 * q] = v.x; Bl[4 * q + 1] = v.y; Bl[4 * q + 2] = v.z; Bl[4 * q + 3] = v.w;
        }
#pragma unroll
        for (int n = 0; n < DSTATE; ++n) {
            float a = expf(dt * A[n]);
            P[n] *= a;
            h[n]  = a * h[n] + du * Bl[n];
        }
    }

    float* Pp = Pout + (size_t)idx * DSTATE;
    float* Hp = Hend + (size_t)idx * DSTATE;
#pragma unroll
    for (int n = 0; n < DSTATE; ++n) { Pp[n] = P[n]; Hp[n] = h[n]; }
}

// =====================================================================
// Scan pass B: sequential carry combine across the 32 chunks per (b,d):
//   h_init[ch+1] = P[ch]*h_init[ch] + h_end[ch].  Only 1536 threads,
//   32 steps — negligible.
// =====================================================================
__global__ __launch_bounds__(256)
void scan_passB_kernel(const float* __restrict__ P,
                       const float* __restrict__ Hend,
                       float* __restrict__ Hinit)
{
    const int idx = blockIdx.x * 256 + threadIdx.x;     // [0, BB*DMODEL)
    if (idx >= BB * DMODEL) return;
    const int b = idx / DMODEL;
    const int d = idx - b * DMODEL;

    float h[DSTATE];
#pragma unroll
    for (int n = 0; n < DSTATE; ++n) h[n] = 0.0f;

    for (int ch = 0; ch < NCH; ++ch) {
        const size_t unit = ((size_t)(ch * BB + b) * DMODEL + d) * DSTATE;
        float* Hi = Hinit + unit;
#pragma unroll
        for (int n = 0; n < DSTATE; ++n) Hi[n] = h[n];
#pragma unroll
        for (int n = 0; n < DSTATE; ++n)
            h[n] = P[unit + n] * h[n] + Hend[unit + n];
    }
}

// =====================================================================
// Scan pass C: replay each chunk from its corrected initial state and
// emit y[b,l,d] = sum_n h[n]*C[n] + u*Ds[d].
// =====================================================================
__global__ __launch_bounds__(256)
void scan_passC_kernel(const float* __restrict__ delta,
                       const float* __restrict__ X,
                       const float* __restrict__ Bsc,
                       const float* __restrict__ Csc,
                       const float* __restrict__ Alog,
                       const float* __restrict__ Ds,
                       const float* __restrict__ Hinit,
                       float* __restrict__ Y)
{
    const int idx = blockIdx.x * 256 + threadIdx.x;
    const int d   = idx % DMODEL;
    const int t2  = idx / DMODEL;
    const int b   = t2 % BB;
    const int ch  = t2 / BB;

    float A[DSTATE], h[DSTATE];
    const float* Hi = Hinit + (size_t)idx * DSTATE;
#pragma unroll
    for (int n = 0; n < DSTATE; ++n) {
        A[n] = -expf(Alog[d * DSTATE + n]);
        h[n] = Hi[n];
    }
    const float Dsd = Ds[d];

    const int l0 = ch * CHL;
    for (int t = 0; t < CHL; ++t) {
        const int bl = b * LL + l0 + t;
        if (t + 8 < CHL) {
            __builtin_prefetch(&delta[(size_t)(bl + 8) * DMODEL + d], 0, 1);
            __builtin_prefetch(&X[(size_t)(bl + 8) * DMODEL + d], 0, 1);
        }
        const float dt = delta[(size_t)bl * DMODEL + d];
        const float u  = X[(size_t)bl * DMODEL + d];
        const float du = dt * u;
        const float4* Bv4 = reinterpret_cast<const float4*>(Bsc + (size_t)bl * DSTATE);
        const float4* Cv4 = reinterpret_cast<const float4*>(Csc + (size_t)bl * DSTATE);
        float Bl[DSTATE], Cl[DSTATE];
#pragma unroll
        for (int q = 0; q < 4; ++q) {
            float4 vb = Bv4[q], vc = Cv4[q];
            Bl[4*q] = vb.x; Bl[4*q+1] = vb.y; Bl[4*q+2] = vb.z; Bl[4*q+3] = vb.w;
            Cl[4*q] = vc.x; Cl[4*q+1] = vc.y; Cl[4*q+2] = vc.z; Cl[4*q+3] = vc.w;
        }
        float y = 0.0f;
#pragma unroll
        for (int n = 0; n < DSTATE; ++n) {
            float a = expf(dt * A[n]);
            h[n] = a * h[n] + du * Bl[n];
            y   += h[n] * Cl[n];
        }
        Y[(size_t)bl * DMODEL + d] = y + u * Dsd;
    }
}

// =====================================================================
extern "C" void kernel_launch(void* const* d_in, const int* in_sizes, int n_in,
                              void* d_out, int out_size, void* d_ws, size_t ws_size,
                              hipStream_t stream)
{
    const float* x      = (const float*)d_in[0];   // (B, L, 384)
    const float* prompt = (const float*)d_in[1];   // (B, L, 16)
    const float* w1     = (const float*)d_in[2];   // (56, 384)
    const float* w2     = (const float*)d_in[3];   // (384, 24)
    const float* bias   = (const float*)d_in[4];   // (384,)
    const float* alog   = (const float*)d_in[5];   // (384, 16)
    const float* ds     = (const float*)d_in[6];   // (384,)
    float*       y      = (float*)d_out;           // (B, L, 384)

    float* ws = (float*)d_ws;
    size_t o = 0;
    float* dtsr  = ws + o; o += (size_t)MTOT * DTRANK;                 // 16384*24
    float* Bsc   = ws + o; o += (size_t)MTOT * DSTATE;                 // 16384*16
    float* Csc   = ws + o; o += (size_t)MTOT * DSTATE;
    float* delta = ws + o; o += (size_t)MTOT * DMODEL;                 // 16384*384
    const size_t units = (size_t)NCH * BB * DMODEL;                    // 49152
    float* P     = ws + o; o += units * DSTATE;
    float* Hend  = ws + o; o += units * DSTATE;
    float* Hinit = ws + o; o += units * DSTATE;
    (void)ws_size; (void)in_sizes; (void)n_in; (void)out_size;

    proj_gemm_kernel<<<dim3(MTOT / 128, 4, 1), 256, 0, stream>>>(x, w1, prompt, dtsr, Bsc, Csc);
    dt_gemm_kernel  <<<dim3(MTOT / 128, 1, 1), 256, 0, stream>>>(dtsr, w2, bias, delta);
    scan_passA_kernel<<<dim3((unsigned)(units / 256)), 256, 0, stream>>>(delta, x, Bsc, alog, P, Hend);
    scan_passB_kernel<<<dim3((BB * DMODEL + 255) / 256), 256, 0, stream>>>(P, Hend, Hinit);
    scan_passC_kernel<<<dim3((unsigned)(units / 256)), 256, 0, stream>>>(delta, x, Bsc, Csc, alog, ds, Hinit, y);
}
